// DoReFaNet_40956808135202
// MI455X (gfx1250) — 8-bit matrix operand order repaired and checked by reading, compiled, not run on hardware
//
#include <hip/hip_runtime.h>
#include <stdint.h>

typedef int v8i_t __attribute__((ext_vector_type(8)));
typedef int v4i_vs __attribute__((vector_size(16)));

#define EPS_BN 1e-4f

#if defined(__has_builtin)
#if __has_builtin(__builtin_amdgcn_global_load_async_to_lds_b128) && \
    __has_builtin(__builtin_amdgcn_s_wait_asynccnt)
#define HAS_ASYNC_LDS 1
#endif
#endif
#ifndef HAS_ASYNC_LDS
#define HAS_ASYNC_LDS 0
#endif

__device__ __forceinline__ void async_copy_b128(const void* g, void* l) {
#if HAS_ASYNC_LDS
  __builtin_amdgcn_global_load_async_to_lds_b128(
      (__attribute__((address_space(1))) v4i_vs*)g,
      (__attribute__((address_space(3))) v4i_vs*)l, 0, 0);
#else
  *(uint4*)l = *(const uint4*)g;
#endif
}

__device__ __forceinline__ void async_wait_all() {
#if HAS_ASYNC_LDS
  __builtin_amdgcn_s_wait_asynccnt(0);
#endif
}

__global__ void zero_f32_kernel(float* p, int n) {
  int i = blockIdx.x * blockDim.x + threadIdx.x;
  if (i < n) p[i] = 0.f;
}

__global__ void absmean_sum_kernel(const float* __restrict__ w, int n, float* acc) {
  __shared__ float red[256];
  float s = 0.f;
  for (int i = blockIdx.x * blockDim.x + threadIdx.x; i < n; i += gridDim.x * blockDim.x)
    s += fabsf(w[i]);
  red[threadIdx.x] = s;
  __syncthreads();
  for (int st = 128; st > 0; st >>= 1) {
    if (threadIdx.x < st) red[threadIdx.x] += red[threadIdx.x + st];
    __syncthreads();
  }
  if (threadIdx.x == 0) atomicAdd(acc, red[0]);
}

__global__ void weight_sign_transpose_kernel(const float* __restrict__ w,
                                             int8_t* __restrict__ wqt,
                                             int Ktot, int Kp, int Cout) {
  long long idx = (long long)blockIdx.x * blockDim.x + threadIdx.x;
  long long total = (long long)Cout * Kp;
  if (idx >= total) return;
  int co = (int)(idx / Kp);
  int k  = (int)(idx % Kp);
  int8_t q = 0;
  if (k < Ktot) q = (w[(size_t)k * Cout + co] >= 0.f) ? (int8_t)1 : (int8_t)-1;
  wqt[idx] = q;
}

__global__ void stem_conv_kernel(const float* __restrict__ x, const float* __restrict__ w0,
                                 const float* __restrict__ b0, float* __restrict__ out) {
  long long idx = (long long)blockIdx.x * blockDim.x + threadIdx.x;
  const long long total = 64LL * 54 * 54 * 24;
  if (idx >= total) return;
  int c0 = (int)(idx % 24) * 4;
  long long p = idx / 24;
  int ox = (int)(p % 54); p /= 54;
  int oy = (int)(p % 54);
  int b  = (int)(p / 54);
  float a0 = b0[c0 + 0], a1 = b0[c0 + 1], a2 = b0[c0 + 2], a3 = b0[c0 + 3];
  const float* xb = x + (size_t)b * 224 * 224 * 3;
  for (int ky = 0; ky < 12; ++ky) {
    int iy = oy * 4 + ky;
    for (int kx = 0; kx < 12; ++kx) {
      int ix = ox * 4 + kx;
      const float* xp = xb + ((size_t)iy * 224 + ix) * 3;
      const float* wp = w0 + (size_t)(ky * 12 + kx) * 3 * 96 + c0;
      float x0 = xp[0], x1 = xp[1], x2 = xp[2];
      float4 w0v = *(const float4*)(wp);
      float4 w1v = *(const float4*)(wp + 96);
      float4 w2v = *(const float4*)(wp + 192);
      a0 += x0 * w0v.x + x1 * w1v.x + x2 * w2v.x;
      a1 += x0 * w0v.y + x1 * w1v.y + x2 * w2v.y;
      a2 += x0 * w0v.z + x1 * w1v.z + x2 * w2v.z;
      a3 += x0 * w0v.w + x1 * w1v.w + x2 * w2v.w;
    }
  }
  float4 r; r.x = a0; r.y = a1; r.z = a2; r.w = a3;
  *(float4*)(out + (((size_t)b * 54 + oy) * 54 + ox) * 96 + c0) = r;
}

__global__ void dorefa_quant_kernel(const float* __restrict__ in, uint8_t* __restrict__ out,
                                    long long n) {
  long long i = (long long)blockIdx.x * blockDim.x + threadIdx.x;
  if (i >= n) return;
  float v = fminf(fmaxf(in[i], 0.f), 1.f);
  out[i] = (uint8_t)(int)rintf(v * 3.f);
}

__global__ void maxpool_quant_kernel(const float* __restrict__ in, uint8_t* __restrict__ out,
                                     int B_, int H, int W, int C, int HO, int WO, int padlo) {
  long long idx = (long long)blockIdx.x * blockDim.x + threadIdx.x;
  long long total = (long long)B_ * HO * WO * C;
  if (idx >= total) return;
  int c = (int)(idx % C);
  long long p = idx / C;
  int ox = (int)(p % WO); p /= WO;
  int oy = (int)(p % HO);
  int b  = (int)(p / HO);
  float m = -3.4e38f;
  for (int dy = 0; dy < 3; ++dy) {
    int iy = oy * 2 - padlo + dy;
    if (iy < 0 || iy >= H) continue;
    for (int dx = 0; dx < 3; ++dx) {
      int ix = ox * 2 - padlo + dx;
      if (ix < 0 || ix >= W) continue;
      m = fmaxf(m, in[(((size_t)b * H + iy) * W + ix) * C + c]);
    }
  }
  float v = fminf(fmaxf(m, 0.f), 1.f);
  out[idx] = (uint8_t)(int)rintf(v * 3.f);
}

template <int CIN, int KWW, int PAD, bool FASTA>
__global__ void __launch_bounds__(256)
qconv_wmma_kernel(const uint8_t* __restrict__ act, const int8_t* __restrict__ wqt,
                  const float* __restrict__ wsum_p, float inv_wcount,
                  const float* __restrict__ beta, const float* __restrict__ mean,
                  const float* __restrict__ var, float* __restrict__ outf,
                  int H, int W, int Cout) {
  constexpr int KTOT = KWW * KWW * CIN;
  constexpr int KP   = (KTOT + 63) & ~63;

  __shared__ uint32_t ldsA[2][64 * 16];
  __shared__ uint32_t ldsB[2][128 * 16];

  const int tid  = threadIdx.x;
  const int lane = tid & 31;
  const int wave = tid >> 5;
  const int hi   = lane >> 4;
  const int ln   = lane & 15;
  const int m0   = blockIdx.x * 64;
  const int co0  = blockIdx.y * 128;

  const int sa_row  = tid >> 2;
  const int sa_q4   = tid & 3;
  const int sb_n    = tid >> 1;
  const int sb_half = tid & 1;

  int a_b, a_y, a_x;
  {
    int gm = m0 + sa_row;
    int hw = H * W;
    a_b = gm / hw;
    int rem = gm - a_b * hw;
    a_y = rem / W;
    a_x = rem - a_y * W;
  }

  auto stageA = [&](int k0, int buf) {
    uint4 v = {0u, 0u, 0u, 0u};
    if (FASTA) {
      const int cell = k0 / CIN;
      const int ci   = (k0 - cell * CIN) + sa_q4 * 16;
      const int kx   = cell % KWW;
      const int ky   = cell / KWW;
      const int iy   = a_y + ky - PAD;
      const int ix   = a_x + kx - PAD;
      if (iy >= 0 && iy < H && ix >= 0 && ix < W)
        v = *(const uint4*)(act + (((size_t)a_b * H + iy) * W + ix) * CIN + ci);
    } else {
      uint32_t pk[4];
#pragma unroll
      for (int q = 0; q < 4; ++q) {
        uint32_t packed = 0;
#pragma unroll
        for (int t = 0; t < 4; ++t) {
          int k = k0 + sa_q4 * 16 + q * 4 + t;
          uint32_t b = 0;
          if (k < KTOT) {
            int cell = k / CIN;
            int ci   = k - cell * CIN;
            int kx   = cell % KWW;
            int ky   = cell / KWW;
            int iy = a_y + ky - PAD;
            int ix = a_x + kx - PAD;
            if (iy >= 0 && iy < H && ix >= 0 && ix < W)
              b = act[(((size_t)a_b * H + iy) * W + ix) * CIN + ci];
          }
          packed |= b << (8 * t);
        }
        pk[q] = packed;
      }
      v.x = pk[0]; v.y = pk[1]; v.z = pk[2]; v.w = pk[3];
    }
    *(uint4*)&ldsA[buf][sa_row * 16 + sa_q4 * 4] = v;
  };

  auto stageB = [&](int k0, int buf) {
    const int8_t* row = wqt + (size_t)(co0 + sb_n) * KP + k0 + sb_half * 32;
    uint32_t* dst = &ldsB[buf][sb_n * 16 + sb_half * 8];
    async_copy_b128(row, dst);
    async_copy_b128(row + 16, dst + 4);
  };

  v8i_t acc[4];
#pragma unroll
  for (int s = 0; s < 4; ++s) acc[s] = (v8i_t){0, 0, 0, 0, 0, 0, 0, 0};

  stageB(0, 0);
  stageA(0, 0);
  async_wait_all();
  __syncthreads();

  int cur = 0;
  for (int k0 = 0; k0 < KP; k0 += 64) {
    const int nxt = cur ^ 1;
    if (k0 + 64 < KP) {
      stageB(k0 + 64, nxt);
      stageA(k0 + 64, nxt);
    }

    v8i_t bf;
#pragma unroll
    for (int i = 0; i < 8; ++i) {
      int kb = ((i >> 2) << 3) + (i & 3) + (hi << 2);
      bf[i] = (int)ldsB[cur][(wave * 16 + ln) * 16 + kb];
    }
#pragma unroll
    for (int s = 0; s < 4; ++s) {
      v8i_t af;
#pragma unroll
      for (int i = 0; i < 8; ++i) {
        int ka = ((i >> 2) << 3) + (i & 3) + (hi << 2);
        af[i] = (int)ldsA[cur][(s * 16 + ln) * 16 + ka];
      }
      acc[s] = __builtin_amdgcn_wmma_i32_16x16x64_iu8(false, af, true, bf, acc[s],
                                                      false, false);
    }

    async_wait_all();
    __syncthreads();
    cur = nxt;
  }

  const int c = co0 + wave * 16 + ln;
  const float wsc = wsum_p[0] * inv_wcount * (1.f / 3.f);
  const float mu = mean[c];
  const float bt = beta[c];
  const float bs = rsqrtf(var[c] + EPS_BN);
#pragma unroll
  for (int s = 0; s < 4; ++s) {
#pragma unroll
    for (int j = 0; j < 8; ++j) {
      int gm = m0 + s * 16 + j + hi * 8;
      float y = (float)acc[s][j] * wsc;
      outf[(size_t)gm * Cout + c] = (y - mu) * bs + bt;
    }
  }
}

__global__ void __launch_bounds__(256)
dense_softmax_kernel(const float* __restrict__ h, const float* __restrict__ wd,
                     const float* __restrict__ bd, float* __restrict__ out) {
  __shared__ float hrow[4096];
  __shared__ float logits[1000];
  __shared__ float red[256];
  const int r = blockIdx.x, tid = threadIdx.x;

  for (int i = tid; i < 4096; i += 256) {
    float v = h[(size_t)r * 4096 + i];
    hrow[i] = fminf(fmaxf(v, 0.f), 1.f);
  }
  __syncthreads();

  for (int c = tid; c < 1000; c += 256) {
    float s = bd[c];
    for (int k = 0; k < 4096; ++k) s += hrow[k] * wd[(size_t)k * 1000 + c];
    logits[c] = s;
  }
  __syncthreads();

  float lm = -3.4e38f;
  for (int c = tid; c < 1000; c += 256) lm = fmaxf(lm, logits[c]);
  red[tid] = lm;
  __syncthreads();
  for (int st = 128; st > 0; st >>= 1) {
    if (tid < st) red[tid] = fmaxf(red[tid], red[tid + st]);
    __syncthreads();
  }
  float mx = red[0];
  __syncthreads();

  float ls = 0.f;
  for (int c = tid; c < 1000; c += 256) {
    float e = __expf(logits[c] - mx);
    logits[c] = e;
    ls += e;
  }
  red[tid] = ls;
  __syncthreads();
  for (int st = 128; st > 0; st >>= 1) {
    if (tid < st) red[tid] += red[tid + st];
    __syncthreads();
  }
  float inv = 1.f / red[0];
  __syncthreads();

  for (int c = tid; c < 1000; c += 256) out[(size_t)r * 1000 + c] = logits[c] * inv;
}

extern "C" void kernel_launch(void* const* d_in, const int* in_sizes, int n_in,
                              void* d_out, int out_size, void* d_ws, size_t ws_size,
                              hipStream_t stream) {
  (void)in_sizes; (void)n_in; (void)out_size; (void)ws_size;
  const float* x     = (const float*)d_in[0];
  const float* w0    = (const float*)d_in[1];
  const float* b0    = (const float*)d_in[2];
  const float* w1    = (const float*)d_in[3];
  const float* beta1 = (const float*)d_in[4];
  const float* mean1 = (const float*)d_in[5];
  const float* var1  = (const float*)d_in[6];
  const float* w2    = (const float*)d_in[7];
  const float* beta2 = (const float*)d_in[8];
  const float* mean2 = (const float*)d_in[9];
  const float* var2  = (const float*)d_in[10];
  const float* w3    = (const float*)d_in[11];
  const float* beta3 = (const float*)d_in[12];
  const float* mean3 = (const float*)d_in[13];
  const float* var3  = (const float*)d_in[14];
  const float* w4    = (const float*)d_in[15];
  const float* beta4 = (const float*)d_in[16];
  const float* mean4 = (const float*)d_in[17];
  const float* var4  = (const float*)d_in[18];
  const float* wf1   = (const float*)d_in[19];
  const float* beta5 = (const float*)d_in[20];
  const float* mean5 = (const float*)d_in[21];
  const float* var5  = (const float*)d_in[22];
  const float* wf2   = (const float*)d_in[23];
  const float* beta6 = (const float*)d_in[24];
  const float* mean6 = (const float*)d_in[25];
  const float* var6  = (const float*)d_in[26];
  const float* wd    = (const float*)d_in[27];
  const float* bd    = (const float*)d_in[28];
  float* out = (float*)d_out;

  uint8_t* ws = (uint8_t*)d_ws;
  size_t off = 0;
  auto carve = [&](size_t bytes) -> size_t {
    size_t o = off;
    off += (bytes + 255) & ~(size_t)255;
    return o;
  };

  float*   scales = (float*)(ws + carve(128));
  int8_t*  q1 = (int8_t*)(ws + carve(256ULL * 2432));
  int8_t*  q2 = (int8_t*)(ws + carve(384ULL * 2304));
  int8_t*  q3 = (int8_t*)(ws + carve(384ULL * 3456));
  int8_t*  q4 = (int8_t*)(ws + carve(256ULL * 3456));
  int8_t*  q5 = (int8_t*)(ws + carve(4096ULL * 9216));
  int8_t*  q6 = (int8_t*)(ws + carve(4096ULL * 4096));
  uint8_t* Aa = ws + carve(17915904ULL);
  uint8_t* Ab = ws + carve(11943936ULL);
  float*   F  = (float*)(ws + carve(191102976ULL));

  auto gb = [](long long n) { return (unsigned)((n + 255) / 256); };

  zero_f32_kernel<<<1, 32, 0, stream>>>(scales, 32);
  absmean_sum_kernel<<<1024, 256, 0, stream>>>(w1,  5*5*96*256,  &scales[0]);
  absmean_sum_kernel<<<1024, 256, 0, stream>>>(w2,  3*3*256*384, &scales[1]);
  absmean_sum_kernel<<<1024, 256, 0, stream>>>(w3,  3*3*384*384, &scales[2]);
  absmean_sum_kernel<<<1024, 256, 0, stream>>>(w4,  3*3*384*256, &scales[3]);
  absmean_sum_kernel<<<2048, 256, 0, stream>>>(wf1, 9216*4096,   &scales[4]);
  absmean_sum_kernel<<<2048, 256, 0, stream>>>(wf2, 4096*4096,   &scales[5]);
  weight_sign_transpose_kernel<<<gb(256LL*2432),  256, 0, stream>>>(w1,  q1, 2400, 2432, 256);
  weight_sign_transpose_kernel<<<gb(384LL*2304),  256, 0, stream>>>(w2,  q2, 2304, 2304, 384);
  weight_sign_transpose_kernel<<<gb(384LL*3456),  256, 0, stream>>>(w3,  q3, 3456, 3456, 384);
  weight_sign_transpose_kernel<<<gb(256LL*3456),  256, 0, stream>>>(w4,  q4, 3456, 3456, 256);
  weight_sign_transpose_kernel<<<gb(4096LL*9216), 256, 0, stream>>>(wf1, q5, 9216, 9216, 4096);
  weight_sign_transpose_kernel<<<gb(4096LL*4096), 256, 0, stream>>>(wf2, q6, 4096, 4096, 4096);

  stem_conv_kernel<<<gb(64LL*54*54*24), 256, 0, stream>>>(x, w0, b0, F);
  dorefa_quant_kernel<<<gb(17915904LL), 256, 0, stream>>>(F, Aa, 17915904LL);

  qconv_wmma_kernel<96, 5, 2, false><<<dim3(186624/64, 2), 256, 0, stream>>>(
      Aa, q1, &scales[0], 1.f/(5.f*5.f*96.f*256.f), beta1, mean1, var1, F, 54, 54, 256);
  maxpool_quant_kernel<<<gb(64LL*27*27*256), 256, 0, stream>>>(F, Ab, 64, 54, 54, 256, 27, 27, 0);

  qconv_wmma_kernel<256, 3, 1, true><<<dim3(46656/64, 3), 256, 0, stream>>>(
      Ab, q2, &scales[1], 1.f/(3.f*3.f*256.f*384.f), beta2, mean2, var2, F, 27, 27, 384);
  maxpool_quant_kernel<<<gb(64LL*14*14*384), 256, 0, stream>>>(F, Aa, 64, 27, 27, 384, 14, 14, 1);

  qconv_wmma_kernel<384, 3, 1, true><<<dim3(12544/64, 3), 256, 0, stream>>>(
      Aa, q3, &scales[2], 1.f/(3.f*3.f*384.f*384.f), beta3, mean3, var3, F, 14, 14, 384);
  dorefa_quant_kernel<<<gb(4816896LL), 256, 0, stream>>>(F, Ab, 4816896LL);

  qconv_wmma_kernel<384, 3, 1, true><<<dim3(12544/64, 2), 256, 0, stream>>>(
      Ab, q4, &scales[3], 1.f/(3.f*3.f*384.f*256.f), beta4, mean4, var4, F, 14, 14, 256);
  maxpool_quant_kernel<<<gb(64LL*6*6*256), 256, 0, stream>>>(F, Aa, 64, 14, 14, 256, 6, 6, 0);

  qconv_wmma_kernel<9216, 1, 0, true><<<dim3(1, 32), 256, 0, stream>>>(
      Aa, q5, &scales[4], 1.f/(9216.f*4096.f), beta5, mean5, var5, F, 1, 1, 4096);
  dorefa_quant_kernel<<<gb(262144LL), 256, 0, stream>>>(F, Ab, 262144LL);

  qconv_wmma_kernel<4096, 1, 0, true><<<dim3(1, 32), 256, 0, stream>>>(
      Ab, q6, &scales[5], 1.f/(4096.f*4096.f), beta6, mean6, var6, F, 1, 1, 4096);

  dense_softmax_kernel<<<64, 256, 0, stream>>>(F, wd, bd, out);
}
